// SpatialAttn_59210419142958
// MI455X (gfx1250) — matrix operand order repaired and checked by reading, not compiled, not run on hardware
//
#include <hip/hip_runtime.h>
#include <hip/hip_bf16.h>
#include <math.h>

// ---------------- problem constants (match reference) ----------------
#define Bc    8
#define Lc    288
#define Nc    400        // nodes
#define PASTc 256        // window
#define Hc    4          // heads
#define Dc    64         // per-head hidden
#define Tc    (Lc - PASTc)   // 32 sliding windows
#define BTc   (Bc * Tc)      // 256

typedef __attribute__((ext_vector_type(16))) _Float16 v16h;
typedef __attribute__((ext_vector_type(8)))  float    v8f;
typedef __attribute__((ext_vector_type(4)))  unsigned int v4u;
typedef __attribute__((ext_vector_type(8)))  int v8i;
typedef __attribute__((ext_vector_type(4)))  int v4i;

#if defined(__has_builtin)
#if __has_builtin(__builtin_amdgcn_tensor_load_to_lds) && \
    __has_builtin(__builtin_amdgcn_s_wait_tensorcnt)
#define HAVE_TDM 1
#endif
#endif
#ifndef HAVE_TDM
#define HAVE_TDM 0
#endif

// =====================================================================
// WMMA fragment helpers — CDNA5 ISA 7.12.2 VGPR layouts (wave32).
// All operand sources are laid out so each lane reads two runs of 8 CONTIGUOUS
// halves (vectorizes to ds_load_b128), no per-element packing.
//
// A (16x32 f16): lane m / m+16 hold row m;
//   lane<16:  halves 0..7 -> K 0..7,  halves 8..15 -> K 16..23
//   lane>=16: halves 0..7 -> K 8..15, halves 8..15 -> K 24..31
// B (32x16 f16): column = lane&15; same K order as A (lane<16: K 0..7,16..23; lane>=16: K 8..15,24..31)
// C/D (16x16 f32): lane<16 -> (M=i,N=lane); lane>=16 -> (M=8+i,N=lane-16)
// =====================================================================

__device__ __forceinline__ v16h load_a_f16(const _Float16* p0, int row0, int k0, int stride) {
  const int lane = threadIdx.x & 31;
  const int m    = lane & 15;
  const int klo  = (lane < 16) ? 0 : 8;
  const _Float16* p = p0 + (row0 + m) * stride + k0 + klo;
  v16h a;
#pragma unroll
  for (int j = 0; j < 8; ++j) { a[j] = p[j]; a[8 + j] = p[16 + j]; }
  return a;
}

// B[k][n] = base[n*stride + k]  (operand stored "n-major": k-runs contiguous)
__device__ __forceinline__ v16h load_b_nmajor(const _Float16* base, int k0, int n0,
                                              int stride) {
  const int lane = threadIdx.x & 31;
  const int n    = lane & 15;
  const int kb   = (lane < 16) ? 0 : 8;
  const _Float16* p = base + (n0 + n) * stride + k0 + kb;
  v16h b;
#pragma unroll
  for (int j = 0; j < 8; ++j) { b[j] = p[j]; b[8 + j] = p[16 + j]; }
  return b;
}

__device__ __forceinline__ void store_cd_f16(_Float16* dst, const v8f& c, int row0,
                                             int col0, int stride) {
  const int lane = threadIdx.x & 31;
  const int n    = lane & 15;
  const int mb   = (lane < 16) ? 0 : 8;
#pragma unroll
  for (int i = 0; i < 8; ++i)
    dst[(size_t)(row0 + mb + i) * stride + col0 + n] = (_Float16)c[i];
}

__device__ __forceinline__ void store_cd_f32(float* dst, const v8f& c, int row0,
                                             int col0, int stride) {
  const int lane = threadIdx.x & 31;
  const int n    = lane & 15;
  const int mb   = (lane < 16) ? 0 : 8;
#pragma unroll
  for (int i = 0; i < 8; ++i)
    dst[(row0 + mb + i) * stride + col0 + n] = c[i];
}

__device__ __forceinline__ void accum_cd_f32(float* dst, const v8f& c, int row0,
                                             int col0, int stride, float scale) {
  const int lane = threadIdx.x & 31;
  const int n    = lane & 15;
  const int mb   = (lane < 16) ? 0 : 8;
#pragma unroll
  for (int i = 0; i < 8; ++i)
    dst[(row0 + mb + i) * stride + col0 + n] += c[i] * scale;
}

#define WMMA_F16(a, b, c) \
  __builtin_amdgcn_wmma_f32_16x16x32_f16(false, (a), false, (b), (short)0, (c), false, false)

// =====================================================================
// Tensor Data Mover: 2D row-major tile -> LDS with per-row padding.
// D# per ISA ch.8: group0 = {flags, lds_addr, gaddr_lo, gaddr_hi|type=2};
// group1 packs data_size/pad + tensor/tile dims + dim0 stride.
// pad_interval code 4 => every 32 DWORDs (one 128B row of 64 halves),
// pad_amount  code 3 => 4 DWORDs (16B)  => 144B LDS row stride.
// This toolchain (clang-23 / therock) uses the 6-arg builtin:
//   (uint32x4 g0, int32x8 g1, int32x4 g2, int32x4 g3, int32x8 gx, i32 cpol)
// groups 2/3 are NULL-equivalent (zeros) for <=2D tensors.
// =====================================================================
#if HAVE_TDM
__device__ __forceinline__ void tdm_load_2d_f16(
    unsigned lds_off, const void* gptr,
    unsigned width_elems, unsigned rows, unsigned row_stride_elems,
    unsigned pad_interval_code, unsigned pad_amount_code, int pad_enable)
{
  unsigned long long ga = (unsigned long long)gptr;
  v4u g0;
  g0.x = 1u;                                        // count=1 (valid), user mode
  g0.y = lds_off;                                   // lds_addr (bytes)
  g0.z = (unsigned)(ga & 0xffffffffu);              // global_addr[31:0]
  g0.w = (unsigned)((ga >> 32) & 0x1ffffffu) | (2u << 30);  // addr[56:32] | type=2

  unsigned d0 = (1u << 16);                         // data_size = 1 (2 bytes)
  if (pad_enable)
    d0 |= (1u << 20) | (pad_interval_code << 22) | (pad_amount_code << 25);
  v8i g1;
  g1[0] = (int)d0;                                  // wg_mask=0 | flags
  g1[1] = (int)((width_elems & 0xffffu) << 16);     // bar_addr=0 | tensor_dim0 lo
  g1[2] = (int)(((width_elems >> 16) & 0xffffu) | ((rows & 0xffffu) << 16)); // dim0 hi | dim1 lo
  g1[3] = (int)(((rows >> 16) & 0xffffu) | ((width_elems & 0xffffu) << 16)); // dim1 hi | tile_dim0
  g1[4] = (int)(rows & 0xffffu);                    // tile_dim1 | tile_dim2=0 (2D)
  g1[5] = (int)(row_stride_elems);                  // tensor_dim0_stride lo
  g1[6] = 0;                                        // stride hi | dim1_stride lo (unused 2D)
  g1[7] = 0;
  v4i z4 = {0, 0, 0, 0};
  v8i z8 = {0, 0, 0, 0, 0, 0, 0, 0};
  __builtin_amdgcn_tensor_load_to_lds(g0, g1, z4, z4, z8, 0);
}
#endif

// =====================================================================
// Kernel 1: q/k/v projection.  One workgroup per (b*t, head).
//   LDS: window x [400][264] f16 (211 KB) + transposed weights
//        Wt [3][64][264] f16 (101 KB)  -> 312.6 KB of 320 KB WGP LDS.
//   300 output tiles (3 mats x 25 x 4) over 8 waves; K=256 -> 8 WMMAs.
//   qkv workspace: [3][BT][H][400][64] f16.
// =====================================================================
#define XS  (PASTc + 8)   // 264 halves, x row stride
#define WTS (PASTc + 8)   // 264 halves, Wt row stride (rows are d)

__global__ __launch_bounds__(256) void qkv_project_kernel(
    const float* __restrict__ inp,  // [B][L][N]
    const float* __restrict__ Wq,   // [H][PAST][D]
    const float* __restrict__ Wk,
    const float* __restrict__ Wv,
    _Float16* __restrict__ qkv)     // [3][BT][H][N][D]
{
  extern __shared__ char smem[];
  _Float16* x_lds = (_Float16*)smem;            // [Nc][XS]
  _Float16* wt    = x_lds + Nc * XS;            // [3][Dc][WTS]  (W transposed)

  const int bt  = blockIdx.x;
  const int b   = bt / Tc;
  const int t   = bt - b * Tc;
  const int h   = blockIdx.y;
  const int tid = threadIdx.x;

  // Stage window: x[n][p] = inp[b, t+p, n]; global reads coalesced over n.
  for (int i = tid; i < Nc * PASTc; i += blockDim.x) {
    int p = i / Nc;
    int n = i - p * Nc;
    x_lds[n * XS + p] = (_Float16)inp[((size_t)(b * Lc + t + p)) * Nc + n];
  }
  // Stage this head's weights TRANSPOSED: wt[mat][d][p] = W[mat][p][d].
  const float* w0 = Wq + (size_t)h * PASTc * Dc;
  const float* w1 = Wk + (size_t)h * PASTc * Dc;
  const float* w2 = Wv + (size_t)h * PASTc * Dc;
  for (int i = tid; i < PASTc * Dc; i += blockDim.x) {
    int p = i >> 6;         // /Dc
    int d = i & 63;
    wt[(0 * Dc + d) * WTS + p] = (_Float16)w0[i];
    wt[(1 * Dc + d) * WTS + p] = (_Float16)w1[i];
    wt[(2 * Dc + d) * WTS + p] = (_Float16)w2[i];
  }
  __syncthreads();

  const int wave = tid >> 5;
  const size_t MATSZ = (size_t)BTc * Hc * Nc * Dc;

  for (int tile = wave; tile < 3 * 25 * 4; tile += 8) {
    const int mat = tile / 100;
    const int r   = tile - mat * 100;
    const int mt  = r >> 2;      // node row tile
    const int nt  = r & 3;       // d col tile
    const _Float16* wb = wt + (size_t)mat * Dc * WTS;

    v8f acc = {};
#pragma unroll
    for (int kk = 0; kk < PASTc / 32; ++kk) {
      v16h a  = load_a_f16(x_lds, mt * 16, kk * 32, XS);
      v16h bf = load_b_nmajor(wb, kk * 32, nt * 16, WTS);   // B[k=p][n=d]=Wt[d][p]
      acc = WMMA_F16(a, bf, acc);
    }
    _Float16* dst = qkv + (size_t)mat * MATSZ + ((size_t)bt * Hc + h) * (Nc * Dc);
    store_cd_f16(dst, acc, mt * 16, nt * 16, Dc);
  }
}

// =====================================================================
// Kernel 2: graph attention.  One workgroup per (b*t, 16-node row tile).
//   Per head: TDM stages k [400][72] and the q row tile into LDS
//   (s_wait_tensorcnt), v is staged transposed vt[64][424]; scores via
//   WMMA, masked softmax (wave16 shuffles) writes f16 attn + f32
//   head-mean; ctx = attn . v via WMMA with zero-padded K tail.
//   Both outputs streamed to HBM exactly once.
// =====================================================================
#define KSTR 72           // halves: k/q row stride (matches TDM pad)
#define VTS  424          // halves: vt row stride (cols = m, padded past 416)
#define ATS  424          // halves: attn16 row stride
#define SSTR 404          // floats: raw scores
#define CSTR 68           // floats: ctx accumulator

// LDS byte offsets (single source of truth; also used in TDM descriptors)
#define K_OFF   0u
#define Q_OFF   (K_OFF + Nc * KSTR * 2u)            // 57600
#define VT_OFF  (Q_OFF + 16u * KSTR * 2u)           // 59904
#define AT_OFF  (VT_OFF + (unsigned)Dc * VTS * 2u)  // 114176
#define S_OFF   (AT_OFF + 16u * ATS * 2u)           // 127744
#define AA_OFF  (S_OFF + 16u * SSTR * 4u)           // 153600
#define CC_OFF  (AA_OFF + 16u * SSTR * 4u)          // 179456
#define LDS2_SZ (CC_OFF + 16u * CSTR * 4u)          // 183808 bytes

__global__ __launch_bounds__(256) void graph_attn_kernel(
    const _Float16* __restrict__ qkv,   // [3][BT][H][N][D]
    const int* __restrict__ mask,       // [N][N]
    float* __restrict__ out,            // [BT][D][N]
    float* __restrict__ attn_out)       // [BT][N][N]
{
  extern __shared__ char smem[];
  _Float16* k_lds  = (_Float16*)(smem + K_OFF);   // [Nc][KSTR]
  _Float16* q_lds  = (_Float16*)(smem + Q_OFF);   // [16][KSTR]
  _Float16* vt     = (_Float16*)(smem + VT_OFF);  // [Dc][VTS]  v transposed
  _Float16* attn16 = (_Float16*)(smem + AT_OFF);  // [16][ATS]  f16 attn
  float*    s_lds  = (float*)(smem + S_OFF);      // [16][SSTR] raw scores
  float*    aacc   = (float*)(smem + AA_OFF);     // [16][SSTR] head-mean attn
  float*    cacc   = (float*)(smem + CC_OFF);     // [16][CSTR] head-mean ctx

  const int bt   = blockIdx.x;
  const int n0   = blockIdx.y * 16;
  const int tid  = threadIdx.x;
  const int wave = tid >> 5;
  const size_t MATSZ = (size_t)BTc * Hc * Nc * Dc;

  for (int i = tid; i < 16 * SSTR; i += blockDim.x) aacc[i] = 0.f;
  for (int i = tid; i < 16 * CSTR; i += blockDim.x) cacc[i] = 0.f;
  // zero-pad ragged K tail once (cols 400..423) -> no guards in ctx GEMM
  for (int i = tid; i < Dc * (VTS - Nc); i += blockDim.x) {
    int d = i / (VTS - Nc), c = Nc + i % (VTS - Nc);
    vt[d * VTS + c] = (_Float16)0.f;
  }
  for (int i = tid; i < 16 * (ATS - Nc); i += blockDim.x) {
    int r = i / (ATS - Nc), c = Nc + i % (ATS - Nc);
    attn16[r * ATS + c] = (_Float16)0.f;
  }

  for (int h = 0; h < Hc; ++h) {
    const _Float16* qg = qkv + ((size_t)bt * Hc + h) * (Nc * Dc);
    const _Float16* kg = qg + MATSZ;
    const _Float16* vg = kg + MATSZ;

    __syncthreads();   // previous head's LDS consumers done / zeros visible

#if HAVE_TDM
    if (tid == 0) {
      // k: 400 rows x 64 halves, row-padded to 72 halves (144B) in LDS
      tdm_load_2d_f16(K_OFF, kg, Dc, Nc, Dc, 4, 3, 1);
      // q tile: 16 rows x 64 halves, same padding
      tdm_load_2d_f16(Q_OFF, qg + (size_t)n0 * Dc, Dc, 16, Dc, 4, 3, 1);
    }
#else
    for (int i = tid; i < Nc * Dc; i += blockDim.x) {
      int n = i >> 6, d = i & 63;
      k_lds[n * KSTR + d] = kg[i];
    }
    for (int i = tid; i < 16 * Dc; i += blockDim.x) {
      int n = i >> 6, d = i & 63;
      q_lds[n * KSTR + d] = qg[(size_t)(n0 + n) * Dc + d];
    }
#endif
    // v staged TRANSPOSED (vt[d][m] = v[m][d]) so ctx B-frags are contiguous
    for (int i = tid; i < Nc * Dc; i += blockDim.x) {
      int n = i >> 6, d = i & 63;
      vt[d * VTS + n] = vg[i];
    }
    if (h + 1 < Hc) {   // warm GL2 for next head (global_prefetch_b8)
      const _Float16* nxt = qg + (size_t)(Nc * Dc);
      __builtin_prefetch(nxt + (tid << 5), 0, 1);
      __builtin_prefetch(nxt + MATSZ + (tid << 5), 0, 1);
      __builtin_prefetch(nxt + 2 * MATSZ + (tid << 5), 0, 1);
    }
#if HAVE_TDM
    __builtin_amdgcn_s_wait_tensorcnt(0);   // issuing wave drains TENSORcnt
#endif
    __syncthreads();

    // ---- scores: S[16 x 400] = q_tile . k^T ; 25 col tiles over 8 waves
    for (int ct = wave; ct < 25; ct += 8) {
      v8f acc = {};
#pragma unroll
      for (int kk = 0; kk < Dc / 32; ++kk) {
        v16h a  = load_a_f16(q_lds, 0, kk * 32, KSTR);
        v16h bf = load_b_nmajor(k_lds, kk * 32, ct * 16, KSTR);  // B[k=d][n=m]=k[m][d]
        acc = WMMA_F16(a, bf, acc);
      }
      store_cd_f32(s_lds, acc, 0, ct * 16, SSTR);
    }
    __syncthreads();

    // ---- mask + softmax; 16 threads per row, wave16 shuffle reductions
    {
      const int row  = tid >> 4;
      const int j    = tid & 15;
      const int grow = n0 + row;
      float vals[25];
      float mx = -3.4e38f;
#pragma unroll
      for (int c = 0; c < 25; ++c) {
        int m   = c * 16 + j;
        float s = s_lds[row * SSTR + m] * 0.125f;            // 1/sqrt(D)
        s = (mask[(size_t)grow * Nc + m] > 0) ? s : -1e9f;   // adjacency
        vals[c] = s;
        mx = fmaxf(mx, s);
      }
#pragma unroll
      for (int o = 8; o >= 1; o >>= 1) mx = fmaxf(mx, __shfl_xor(mx, o, 16));
      float sum = 0.f;
#pragma unroll
      for (int c = 0; c < 25; ++c) { float e = __expf(vals[c] - mx); vals[c] = e; sum += e; }
#pragma unroll
      for (int o = 8; o >= 1; o >>= 1) sum += __shfl_xor(sum, o, 16);
      const float inv = 1.f / sum;
#pragma unroll
      for (int c = 0; c < 25; ++c) {
        int m = c * 16 + j;
        float a = vals[c] * inv;
        attn16[row * ATS + m] = (_Float16)a;        // f16 copy feeds ctx GEMM
        aacc[row * SSTR + m] += a * (1.f / Hc);     // head-mean attention
      }
    }
    __syncthreads();

    // ---- ctx[16 x 64] += attn[16 x 400] . v[400 x 64]; waves 0..3
    if (wave < 4) {
      const int nt = wave;
      v8f acc = {};
#pragma unroll
      for (int kt = 0; kt < 13; ++kt) {   // 13*32 = 416, tail zero-padded
        v16h a  = load_a_f16(attn16, 0, kt * 32, ATS);
        v16h bf = load_b_nmajor(vt, kt * 32, nt * 16, VTS);  // B[k=m][n=d]=vt[d][m]
        acc = WMMA_F16(a, bf, acc);
      }
      accum_cd_f32(cacc, acc, 0, nt * 16, CSTR, 1.f / Hc);
    }
  }
  __syncthreads();

  // ---- single streaming store of both outputs
  for (int i = tid; i < 16 * Dc; i += blockDim.x) {
    int r = i >> 6, d = i & 63;
    out[((size_t)bt * Dc + d) * Nc + (n0 + r)] = cacc[r * CSTR + d];
  }
  for (int i = tid; i < 16 * Nc; i += blockDim.x) {
    int r = i / Nc, m = i - r * Nc;
    attn_out[((size_t)bt * Nc + (n0 + r)) * Nc + m] = aacc[r * SSTR + m];
  }
}

// =====================================================================
// Host launcher
// =====================================================================
extern "C" void kernel_launch(void* const* d_in, const int* in_sizes, int n_in,
                              void* d_out, int out_size, void* d_ws, size_t ws_size,
                              hipStream_t stream) {
  const float* inp  = (const float*)d_in[0];   // [B][L][N] f32
  const int*   mask = (const int*)d_in[1];     // [N][N] i32
  const float* Wq   = (const float*)d_in[2];   // [H][PAST][D] f32
  const float* Wk   = (const float*)d_in[3];
  const float* Wv   = (const float*)d_in[4];

  _Float16* qkv = (_Float16*)d_ws;             // [3][BT][H][N][D] f16 (157 MB)

  float* out      = (float*)d_out;                   // [BT][D][N]
  float* attn_out = out + (size_t)BTc * Dc * Nc;     // [BT][N][N]

  const size_t lds1 = (size_t)(Nc * XS + 3 * Dc * WTS) * sizeof(_Float16);
  qkv_project_kernel<<<dim3(BTc, Hc), 256, lds1, stream>>>(inp, Wq, Wk, Wv, qkv);

  graph_attn_kernel<<<dim3(BTc, Nc / 16), 256, (size_t)LDS2_SZ, stream>>>(
      qkv, mask, out, attn_out);
}
